// NeuromorphicLM_79731772883719
// MI455X (gfx1250) — matrix operand order repaired and checked by reading, not compiled, not run on hardware
//
#include <hip/hip_runtime.h>
#include <hip/hip_bf16.h>
#include <math.h>

// ---- model dims ----
#define BSZ    8
#define SEQ    256
#define VOCAB  32000
#define DMODEL 768
#define NB     4          // B memory blocks
#define NC     8          // C columns per block
#define DCOL   64
#define DMEM   64
#define GG     32         // G = NB*NC
#define GD     2048       // G*DCOL
#define RS     32         // PM slots
#define MEM    256        // EM slots
#define CEM    8          // EM write candidates
#define HHID   64         // neuromodulator hidden
#define TT     (BSZ*SEQ)  // 2048 tokens
#define PM_DECAY 0.99f
#define AGE_FACTOR 0.93941306f   // exp(-256/4096)

typedef __attribute__((ext_vector_type(16))) _Float16 v16h;
typedef __attribute__((ext_vector_type(8)))  float    v8f;

__device__ __forceinline__ float sigmoidf_(float x){ return 1.f/(1.f+expf(-x)); }
__device__ __forceinline__ float softplusf_(float x){ return (x>20.f)? x : log1pf(expf(x)); }

// ============================================================================
// Tiled WMMA GEMM: C[M,N] = A[M,K] @ B[K,N] (+bias[n]).
// K, N are compile-time so per-element strides fold into the 24-bit global
// IOFFSET immediates (no 64-bit address chains in the staging loops).
// TRANSB: B element (k,n) comes from Bm[n*K + k]  (used for logits = xf @ emb^T)
//
// Block = 128 threads (4 waves), tile 64x64, K-step 32.
//  - A: gathered straight into per-wave register fragments (wave w owns rows
//    16w..16w+15). Per lane: one 16-float K-contiguous run; the two half-lanes
//    of a row jointly cover the full 128B line.
//  - B: each thread gathers one 16-half fragment slot from global, converts to
//    f16, stores one v16h to fragment-major LDS (2x ds_store_b128). Each wave
//    then reads the 4 B fragments as whole v16h's (2x ds_load_b128 each).
// A-fragment lane layout (16-bit A 16x32): lanes 0-15 K 0..15,
// lanes 16-31 K 16..31. B lane layout: lanes 0-15 K 0..15, 16-31 K 16..31.
// ============================================================================
template<int K, int N, bool TRANSB, bool BIAS>
__global__ __launch_bounds__(128) void gemm_wmma_kernel(
    const float* __restrict__ A, const float* __restrict__ Bm,
    const float* __restrict__ bias, float* __restrict__ C)
{
  __shared__ alignas(32) _Float16 sB[4*32*16];   // [frag t][lane][e], 4KB
  const int tid  = threadIdx.x;
  const int lane = tid & 31, wave = tid >> 5;
  const int half = lane >> 4, l16 = lane & 15;
  const int rowBase = blockIdx.y * 64;
  const int colBase = blockIdx.x * 64;

  // this lane's A row (per-wave fragment) and this thread's staged B column
  const float* __restrict__ Arow = A + (size_t)(rowBase + wave*16 + l16)*K;
  const int bN = colBase + wave*16 + l16;        // staging slot: frag t == wave
  // base pointers so inner-loop offsets are compile-time immediates
  const float* __restrict__ BrowT = Bm + (size_t)bN*K;   // TRANSB: + k
  const float* __restrict__ BcolN = Bm + bN;             // !TRANSB: + k*N

  v8f acc[4] = {};

  for (int k0 = 0; k0 < K; k0 += 32) {
    // ---- A fragment: registers, no LDS ----
    v16h a;
    #pragma unroll
    for (int e = 0; e < 16; ++e) {
      int ka = k0 + 16*half + e;
      a[e] = (_Float16)Arow[ka];
    }
    // ---- B fragment slot: gather -> f16 -> one v16h LDS store ----
    v16h tb;
    #pragma unroll
    for (int e = 0; e < 16; ++e) {
      int kb = k0 + 16*half + e;
      float v = TRANSB ? BrowT[kb] : BcolN[(size_t)kb*N];
      tb[e] = (_Float16)v;
    }
    ((v16h*)sB)[tid] = tb;

    if (k0 + 32 < K) {      // prefetch next K-tile (global_prefetch_b8)
      __builtin_prefetch(Arow + k0 + 32, 0, 1);
      __builtin_prefetch(TRANSB ? (BrowT + k0 + 32)
                                : (BcolN + (size_t)(k0 + 32)*N), 0, 1);
    }
    __syncthreads();

    #pragma unroll
    for (int t = 0; t < 4; ++t) {
      v16h b = ((const v16h*)sB)[t*32 + lane];
      acc[t] = __builtin_amdgcn_wmma_f32_16x16x32_f16(
          false, a, false, b, (short)0, acc[t], false, false);
    }
    __syncthreads();
  }

  // C/D layout: lanes 0-15 -> M = v; lanes 16-31 -> M = v+8
  #pragma unroll
  for (int t = 0; t < 4; ++t) {
    #pragma unroll
    for (int v = 0; v < 8; ++v) {
      int r = rowBase + wave*16 + half*8 + v;
      int c = colBase + t*16 + l16;
      float val = acc[t][v];
      if (BIAS) val += bias[c];
      C[(size_t)r*N + c] = val;
    }
  }
}

// ============================================================================
// Embedding gather: x[t,:] = emb[ids[t],:] + pos_emb[t%SEQ,:]
// ============================================================================
__global__ __launch_bounds__(256) void embed_kernel(
    const float* __restrict__ emb, const float* __restrict__ pos,
    const int* __restrict__ ids, float* __restrict__ x)
{
  int t = blockIdx.x, tid = threadIdx.x;
  int n = t % SEQ;
  int id = ids[t];
  for (int d = tid; d < DMODEL; d += 256)
    x[(size_t)t*DMODEL + d] = emb[(size_t)id*DMODEL + d] + pos[(size_t)n*DMODEL + d];
}

// ============================================================================
// Cortical column + PM/EM reads + feedback. One block (64 thr) per (token, group).
// ============================================================================
__global__ __launch_bounds__(64) void columns_kernel(
    const float* __restrict__ xcols, const float* __restrict__ W_col,
    const float* __restrict__ b_col, const float* __restrict__ W_q,
    const float* __restrict__ pmK, const float* __restrict__ pmV,
    const float* __restrict__ emK, const float* __restrict__ emV,
    const float* __restrict__ Wback_pm, const float* __restrict__ Wback_em,
    float* __restrict__ xout)
{
  int g = blockIdx.x, t = blockIdx.y, tid = threadIdx.x;
  int s = t / SEQ, b = g / NC;
  __shared__ float xc[DCOL], h[DCOL], q[DMEM], rpm[DMEM], rem[DMEM];
  __shared__ float sc[MEM];
  __shared__ float red[2];

  size_t base = ((size_t)t*GG + g)*DCOL;
  xc[tid] = xcols[base + tid];
  __syncthreads();

  { // h
    const float* W = W_col + (size_t)g*DCOL*DCOL;
    float acc = b_col[g*DCOL + tid];
    for (int e = 0; e < DCOL; ++e) acc += xc[e]*W[e*DCOL + tid];
    h[tid] = tanhf(acc);
  }
  __syncthreads();
  { // q
    float acc = 0.f;
    for (int d = 0; d < DCOL; ++d) acc += h[d]*W_q[d*DMEM + tid];
    q[tid] = acc;
  }
  __syncthreads();

  // ---- PM attention (32 slots) ----
  const float* pmKb = pmK + (size_t)(s*NB + b)*RS*DMEM;
  const float* pmVb = pmV + (size_t)(s*NB + b)*RS*DMEM;
  if (tid < RS) {
    const float* k = pmKb + tid*DMEM;
    float acc = 0.f;
    for (int m = 0; m < DMEM; ++m) acc += q[m]*k[m];
    sc[tid] = acc * 0.125f;
  }
  __syncthreads();
  if (tid == 0) {
    float mx = sc[0];
    for (int r = 1; r < RS; ++r) mx = fmaxf(mx, sc[r]);
    float su = 0.f;
    for (int r = 0; r < RS; ++r) { sc[r] = expf(sc[r]-mx); su += sc[r]; }
    red[0] = 1.f/su;
  }
  __syncthreads();
  {
    float inv = red[0], acc = 0.f;
    for (int r = 0; r < RS; ++r) acc += sc[r]*inv*pmVb[r*DMEM + tid];
    rpm[tid] = acc;
  }
  __syncthreads();

  // ---- EM attention (256 slots) ----
  const float* emKb = emK + (size_t)(s*NB + b)*MEM*DMEM;
  const float* emVb = emV + (size_t)(s*NB + b)*MEM*DMEM;
  for (int M = tid; M < MEM; M += DCOL) {
    const float* k = emKb + M*DMEM;
    float acc = 0.f;
    for (int m = 0; m < DMEM; ++m) acc += q[m]*k[m];
    sc[M] = acc * 0.125f;
  }
  __syncthreads();
  if (tid == 0) {
    float mx = sc[0];
    for (int M = 1; M < MEM; ++M) mx = fmaxf(mx, sc[M]);
    float su = 0.f;
    for (int M = 0; M < MEM; ++M) { sc[M] = expf(sc[M]-mx); su += sc[M]; }
    red[1] = 1.f/su;
  }
  __syncthreads();
  {
    float inv = red[1], acc = 0.f;
    for (int M = 0; M < MEM; ++M) acc += sc[M]*inv*emVb[M*DMEM + tid];
    rem[tid] = acc;
  }
  __syncthreads();

  { // feedback + output
    float acc = 0.f;
    for (int m = 0; m < DMEM; ++m)
      acc += rpm[m]*Wback_pm[m*DCOL + tid] + rem[m]*Wback_em[m*DCOL + tid];
    xout[base + tid] = tanhf(h[tid] + acc);
  }
}

// ============================================================================
// Candidate heads, gates, prediction, surprise, novelty. One block per (t,g).
// ============================================================================
__global__ __launch_bounds__(64) void heads_kernel(
    const float* __restrict__ xout, const float* __restrict__ W_k,
    const float* __restrict__ W_v, const float* __restrict__ W_q,
    const float* __restrict__ W_vn, const float* __restrict__ w_gate,
    const float* __restrict__ w_wnov, const float* __restrict__ W_pred,
    const float* __restrict__ emK, const float* __restrict__ zhat_prev,
    int first_step,
    float* __restrict__ kc_o, float* __restrict__ vc_o, float* __restrict__ qn_o,
    float* __restrict__ vn_o, float* __restrict__ zh_o,
    float* __restrict__ gate_o, float* __restrict__ nov_o)
{
  int g = blockIdx.x, t = blockIdx.y, tid = threadIdx.x;
  int s = t / SEQ, n = t % SEQ, b = g / NC, c = g % NC;
  __shared__ float z2[DCOL], qn[DMEM], red[DCOL];
  __shared__ float sG[4];   // 0:gate 1:wnov 2:surp 3:||qn||+eps

  size_t base = ((size_t)t*GG + g)*DCOL;
  z2[tid] = xout[base + tid];
  __syncthreads();

  float a0=0,a1=0,a2=0,a3=0,a4=0;
  for (int d = 0; d < DCOL; ++d) {
    float zd = z2[d];
    a0 += zd*W_k [d*DMEM + tid];
    a1 += zd*W_v [d*DMEM + tid];
    a2 += zd*W_q [d*DMEM + tid];
    a3 += zd*W_vn[d*DMEM + tid];
    a4 += zd*W_pred[d*DCOL + tid];
  }
  kc_o[base+tid]=a0; vc_o[base+tid]=a1; qn_o[base+tid]=a2;
  vn_o[base+tid]=a3; zh_o[base+tid]=a4;
  qn[tid]=a2;

  red[tid] = z2[tid]*w_gate[tid];
  __syncthreads();
  if (tid==0){ float sa=0; for(int d=0;d<DCOL;++d) sa+=red[d]; sG[0]=sigmoidf_(sa); }
  __syncthreads();
  red[tid] = z2[tid]*w_wnov[tid];
  __syncthreads();
  if (tid==0){ float sa=0; for(int d=0;d<DCOL;++d) sa+=red[d]; sG[1]=sigmoidf_(sa); }
  __syncthreads();
  if (first_step) {
    if (tid==0) sG[2]=1.f;
  } else {
    float dif = z2[tid]-zhat_prev[base+tid];
    red[tid]=dif*dif;
    __syncthreads();
    if (tid==0){ float sa=0; for(int d=0;d<DCOL;++d) sa+=red[d]; sG[2]=sqrtf(sa); }
  }
  __syncthreads();
  red[tid]=qn[tid]*qn[tid];
  __syncthreads();
  if (tid==0){ float sa=0; for(int d=0;d<DCOL;++d) sa+=red[d]; sG[3]=sqrtf(sa)+1e-8f; }
  __syncthreads();
  if (tid==0) gate_o[(size_t)t*GG + g] = sG[0];

  // max cosine similarity against EM keys
  const float* emKb = emK + (size_t)(s*NB + b)*MEM*DMEM;
  float qinv = 1.f/sG[3];
  float mx = -1e30f;
  for (int M = tid; M < MEM; M += DCOL) {
    const float* k = emKb + M*DMEM;
    float dot=0.f, nk=0.f;
    for (int m=0;m<DMEM;++m){ float kv=k[m]; dot+=qn[m]*kv; nk+=kv*kv; }
    mx = fmaxf(mx, dot*qinv/(sqrtf(nk)+1e-8f));
  }
  red[tid]=mx;
  __syncthreads();
  if (tid==0) {
    float m2=red[0];
    for(int d=1;d<DCOL;++d) m2=fmaxf(m2,red[d]);
    nov_o[(size_t)(s*NB+b)*(SEQ*NC) + n*NC + c] = sG[2]*sG[1]*(1.f - m2);
  }
}

// ============================================================================
// PM soft routing: gw[s,b,item,r] = gate * softmax_r( unit(k_cand) . pm_K[r] )
// ============================================================================
__global__ __launch_bounds__(32) void route_kernel(
    const float* __restrict__ kc, const float* __restrict__ pmK,
    const float* __restrict__ gate, float* __restrict__ gw)
{
  int g = blockIdx.x, t = blockIdx.y, tid = threadIdx.x;
  int s = t/SEQ, n = t%SEQ, b = g/NC, c = g%NC;
  __shared__ float k[DCOL], scs[RS];
  __shared__ float invn, invsum;
  size_t base = ((size_t)t*GG + g)*DCOL;
  k[tid] = kc[base+tid]; k[tid+32] = kc[base+tid+32];
  __syncthreads();
  if (tid==0){ float s2=0; for(int d=0;d<DCOL;++d) s2+=k[d]*k[d]; invn=1.f/(sqrtf(s2)+1e-8f); }
  __syncthreads();
  const float* pk = pmK + ((size_t)(s*NB+b)*RS + tid)*DMEM;
  float dot=0.f;
  for (int d=0; d<DCOL; ++d) dot += k[d]*pk[d];
  scs[tid] = dot*invn;                       // TAU_ROUTE = 1
  __syncthreads();
  if (tid==0){
    float mx=scs[0]; for(int r=1;r<RS;++r) mx=fmaxf(mx,scs[r]);
    float su=0; for(int r=0;r<RS;++r){ scs[r]=expf(scs[r]-mx); su+=scs[r]; }
    invsum=1.f/su;
  }
  __syncthreads();
  float gt = gate[(size_t)t*GG+g];
  gw[((size_t)(s*NB+b)*(SEQ*NC) + n*NC + c)*RS + tid] = gt*scs[tid]*invsum;
}

// ============================================================================
// Eligibility reduction: elig_{K,V}[s,b,r,d] = sum_items gw * cand.
// ============================================================================
__global__ __launch_bounds__(64) void elig_kernel(
    const float* __restrict__ gw, const float* __restrict__ kc,
    const float* __restrict__ vc, float* __restrict__ eligK, float* __restrict__ eligV)
{
  int r = blockIdx.x, sb = blockIdx.y, d = threadIdx.x;
  int s = sb/NB, b = sb%NB;
  float aK=0.f, aV=0.f;
  for (int it = 0; it < SEQ*NC; ++it) {
    int n = it/NC, c = it%NC;
    size_t base = (((size_t)(s*SEQ+n))*GG + b*NC + c)*DCOL;
    float w = gw[((size_t)sb*(SEQ*NC)+it)*RS + r];
    aK += w*kc[base+d];
    aV += w*vc[base+d];
  }
  eligK[((size_t)sb*RS + r)*DMEM + d] = aK;
  eligV[((size_t)sb*RS + r)*DMEM + d] = aV;
}

// ============================================================================
// Top-CEM novelty selection per (s,b): iterative argmax over 2048.
// ============================================================================
__global__ __launch_bounds__(64) void topk_kernel(
    const float* __restrict__ nov, float* __restrict__ tops, int* __restrict__ topi)
{
  int sb = blockIdx.x, tid = threadIdx.x;
  __shared__ float w[SEQ*NC];
  __shared__ float bv[64]; __shared__ int bi[64];
  const float* row = nov + (size_t)sb*(SEQ*NC);
  for (int i=tid; i<SEQ*NC; i+=64) w[i]=row[i];
  __syncthreads();
  for (int j=0; j<CEM; ++j){
    float mv=-1e30f; int mi=0;
    for (int i=tid; i<SEQ*NC; i+=64) if (w[i]>mv){mv=w[i];mi=i;}
    bv[tid]=mv; bi[tid]=mi;
    __syncthreads();
    if (tid==0){
      float m2=bv[0]; int i2=bi[0];
      for(int d=1;d<64;++d) if (bv[d]>m2){m2=bv[d];i2=bi[d];}
      tops[sb*CEM+j]=m2; topi[sb*CEM+j]=i2; w[i2]=-1e30f;
    }
    __syncthreads();
  }
}

// ============================================================================
// PM neuromodulated commit. Block per (s,b).
// ============================================================================
__global__ __launch_bounds__(64) void pm_commit_kernel(
    const float* __restrict__ eligK, const float* __restrict__ eligV,
    const float* __restrict__ W1, const float* __restrict__ b1,
    const float* __restrict__ Wg, const float* __restrict__ Ws,
    const float* __restrict__ Wt,
    float* __restrict__ pmK, float* __restrict__ pmV, float* __restrict__ pma)
{
  int sb = blockIdx.x, tid = threadIdx.x;
  __shared__ float feat[2+DMEM], h[HHID], gwv[RS];
  __shared__ float sg, stau, sinv;
  const float* eK = eligK + (size_t)sb*RS*DMEM;
  const float* eV = eligV + (size_t)sb*RS*DMEM;
  float* pa = pma + sb*RS;

  if (tid < RS) pa[tid] *= PM_DECAY;
  if (tid < RS) {
    float s2=0; for (int d=0;d<DMEM;++d){ float v=eK[tid*DMEM+d]; s2+=v*v; }
    gwv[tid]=sqrtf(s2);
  }
  __syncthreads();
  if (tid==0){
    float nm=0, as=0;
    for (int r=0;r<RS;++r){ nm+=gwv[r]; as+=pa[r]; }
    feat[0]=nm/RS; feat[1]=as;
  }
  {
    float acc=0; for (int r=0;r<RS;++r) acc+=eK[r*DMEM+tid];
    feat[2+tid]=acc/RS;
  }
  __syncthreads();
  {
    float acc=b1[tid];
    for (int f=0; f<2+DMEM; ++f) acc += feat[f]*W1[f*HHID + tid];
    h[tid]=tanhf(acc);
  }
  __syncthreads();
  if (tid==0){
    float ag=0, at=0;
    for (int k=0;k<HHID;++k){ ag+=h[k]*Wg[k]; at+=h[k]*Wt[k]; }
    sg = sigmoidf_(ag);
    stau = softplusf_(at) + 0.1f;
  }
  __syncthreads();
  if (tid<RS){
    float acc=0; for (int k=0;k<HHID;++k) acc+=h[k]*Ws[k*RS+tid];
    gwv[tid]=acc/stau;
  }
  __syncthreads();
  if (tid==0){
    float mx=gwv[0]; for(int r=1;r<RS;++r) mx=fmaxf(mx,gwv[r]);
    float su=0; for(int r=0;r<RS;++r){ gwv[r]=expf(gwv[r]-mx); su+=gwv[r]; }
    sinv = sg/su;
  }
  __syncthreads();
  if (tid<RS){ gwv[tid]*=sinv; pa[tid]+=gwv[tid]; }
  __syncthreads();
  float* pK = pmK + (size_t)sb*RS*DMEM;
  float* pV = pmV + (size_t)sb*RS*DMEM;
  for (int i=tid; i<RS*DMEM; i+=64){
    int r = i/DMEM;
    pK[i] += gwv[r]*eK[i];
    pV[i] += gwv[r]*eV[i];
  }
}

// ============================================================================
// EM neuromodulated decayed write (scatter into least-used slots). Block per (s,b).
// ============================================================================
__global__ __launch_bounds__(64) void em_write_kernel(
    const float* __restrict__ qn, const float* __restrict__ vn,
    const float* __restrict__ tops, const int* __restrict__ topi,
    const float* __restrict__ W1, const float* __restrict__ b1,
    const float* __restrict__ Wg, const float* __restrict__ Wt,
    const float* __restrict__ Wd,
    float* __restrict__ emK, float* __restrict__ emV, float* __restrict__ emS)
{
  int sb = blockIdx.x, tid = threadIdx.x;
  int s = sb/NB, b = sb%NB;
  __shared__ float cK[CEM*DMEM], cV[CEM*DMEM];
  __shared__ float feat[2+DMEM], h[HHID];
  __shared__ float gv, tauv, decv;
  __shared__ float wS[MEM], alpha[CEM];
  __shared__ int wri[CEM];
  __shared__ float bv[64]; __shared__ int bi[64];
  float* eS = emS + sb*MEM;

  for (int i=tid; i<CEM*DMEM; i+=64){
    int j=i/DMEM, m=i%DMEM;
    int it = topi[sb*CEM+j];
    int n = it/NC, c = it%NC;
    size_t base = (((size_t)(s*SEQ+n))*GG + b*NC + c)*DCOL;
    cK[i]=qn[base+m]; cV[i]=vn[base+m];
  }
  __syncthreads();
  if (tid==0){
    float ts=0; for(int j=0;j<CEM;++j) ts+=tops[sb*CEM+j];
    feat[0]=ts/CEM;
    float ss=0; for(int M=0;M<MEM;++M) ss+=eS[M];
    feat[1]=ss;
  }
  { float acc=0; for(int j=0;j<CEM;++j) acc+=cK[j*DMEM+tid]; feat[2+tid]=acc/CEM; }
  __syncthreads();
  { float acc=b1[tid]; for(int f=0;f<2+DMEM;++f) acc+=feat[f]*W1[f*HHID+tid]; h[tid]=tanhf(acc); }
  __syncthreads();
  if (tid==0){
    float ag=0, at=0, ad=0;
    for(int k=0;k<HHID;++k){ ag+=h[k]*Wg[k]; at+=h[k]*Wt[k]; ad+=h[k]*Wd[k]; }
    gv = sigmoidf_(ag);
    tauv = softplusf_(at) + 0.1f;
    decv = sigmoidf_(ad);
  }
  __syncthreads();
  for (int M=tid; M<MEM; M+=64){ eS[M]*=decv; wS[M]=eS[M]; }
  __syncthreads();
  // CEM smallest-scoring slots
  for (int j=0; j<CEM; ++j){
    float mv=1e30f; int mi=0;
    for (int M=tid; M<MEM; M+=64) if (wS[M]<mv){mv=wS[M];mi=M;}
    bv[tid]=mv; bi[tid]=mi;
    __syncthreads();
    if (tid==0){
      float m2=bv[0]; int i2=bi[0];
      for(int d=1;d<64;++d) if (bv[d]<m2){m2=bv[d];i2=bi[d];}
      wri[j]=i2; wS[i2]=1e30f;
    }
    __syncthreads();
  }
  if (tid==0){
    float mx=-1e30f;
    for(int j=0;j<CEM;++j) mx=fmaxf(mx, tops[sb*CEM+j]/tauv);
    float su=0, e[CEM];
    for(int j=0;j<CEM;++j){ e[j]=expf(tops[sb*CEM+j]/tauv - mx); su+=e[j]; }
    for(int j=0;j<CEM;++j) alpha[j]=gv*e[j]/su;
  }
  __syncthreads();
  float* eK = emK + (size_t)sb*MEM*DMEM;
  float* eV = emV + (size_t)sb*MEM*DMEM;
  for (int i=tid; i<CEM*DMEM; i+=64){
    int j=i/DMEM, m=i%DMEM;
    int M = wri[j];
    float a = alpha[j];
    eK[M*DMEM+m] = (1.f-a)*eK[M*DMEM+m] + a*cK[i];
    eV[M*DMEM+m] = (1.f-a)*eV[M*DMEM+m] + a*cV[i];
  }
  __syncthreads();
  if (tid<CEM) eS[wri[tid]] += alpha[tid]*tops[sb*CEM+tid];
  __syncthreads();
  for (int M=tid; M<MEM; M+=64) eS[M]*=AGE_FACTOR;
}

// ============================================================================
// Refinement mix: x_cols = (r==0) ? x_out : (1-lam)*x_cols + lam*x_out
// ============================================================================
__global__ __launch_bounds__(256) void mix_kernel(
    float* __restrict__ xcols, const float* __restrict__ xout,
    const float* __restrict__ lambda_logit, int first)
{
  size_t i = (size_t)blockIdx.x*256 + threadIdx.x;
  if (i >= (size_t)TT*GD) return;
  if (first) { xcols[i]=xout[i]; return; }
  float lam = sigmoidf_(lambda_logit[0]);
  xcols[i] = (1.f-lam)*xcols[i] + lam*xout[i];
}

// ============================================================================
// Layernorm (in place over xf rows of 768)
// ============================================================================
__global__ __launch_bounds__(256) void layernorm_kernel(
    float* __restrict__ xf, const float* __restrict__ gam, const float* __restrict__ bet)
{
  int t = blockIdx.x, tid = threadIdx.x;
  __shared__ float red[256]; __shared__ float mu, rstd;
  float* row = xf + (size_t)t*DMODEL;
  float acc=0;
  for (int d=tid; d<DMODEL; d+=256) acc+=row[d];
  red[tid]=acc; __syncthreads();
  if (tid==0){ float s2=0; for(int k=0;k<256;++k) s2+=red[k]; mu=s2/DMODEL; }
  __syncthreads();
  acc=0;
  for (int d=tid; d<DMODEL; d+=256){ float v=row[d]-mu; acc+=v*v; }
  red[tid]=acc; __syncthreads();
  if (tid==0){ float s2=0; for(int k=0;k<256;++k) s2+=red[k]; rstd=rsqrtf(s2/DMODEL+1e-5f); }
  __syncthreads();
  for (int d=tid; d<DMODEL; d+=256) row[d]=(row[d]-mu)*rstd*gam[d]+bet[d];
}

// ============================================================================
extern "C" void kernel_launch(void* const* d_in, const int* in_sizes, int n_in,
                              void* d_out, int out_size, void* d_ws, size_t ws_size,
                              hipStream_t stream)
{
  (void)in_sizes; (void)n_in; (void)out_size; (void)ws_size;
  const float* emb      = (const float*)d_in[0];
  const float* pos_emb  = (const float*)d_in[1];
  const float* W_fo     = (const float*)d_in[2];
  const float* b_fo     = (const float*)d_in[3];
  const float* W_col    = (const float*)d_in[4];
  const float* b_col    = (const float*)d_in[5];
  const float* W_pred   = (const float*)d_in[6];
  const float* W_k      = (const float*)d_in[7];
  const float* W_v      = (const float*)d_in[8];
  const float* W_q      = (const float*)d_in[9];
  const float* W_vn     = (const float*)d_in[10];
  const float* w_gate   = (const float*)d_in[11];
  const float* w_wnov   = (const float*)d_in[12];
  const float* Wback_pm = (const float*)d_in[13];
  const float* Wback_em = (const float*)d_in[14];
  const float* W_fi     = (const float*)d_in[15];
  const float* b_fi     = (const float*)d_in[16];
  const float* ln_g     = (const float*)d_in[17];
  const float* ln_b     = (const float*)d_in[18];
  const float* lam_l    = (const float*)d_in[19];
  const float* W1_pm    = (const float*)d_in[20];
  const float* b1_pm    = (const float*)d_in[21];
  const float* Wg_pm    = (const float*)d_in[22];
  const float* Ws_pm    = (const float*)d_in[23];
  const float* Wt_pm    = (const float*)d_in[24];
  const float* W1_em    = (const float*)d_in[25];
  const float* b1_em    = (const float*)d_in[26];
  const float* Wg_em    = (const float*)d_in[27];
  const float* Wt_em    = (const float*)d_in[28];
  const float* Wd_em    = (const float*)d_in[29];
  const float* pmK_in   = (const float*)d_in[30];
  const float* pmV_in   = (const float*)d_in[31];
  const float* pma_in   = (const float*)d_in[32];
  const float* emK_in   = (const float*)d_in[33];
  const float* emV_in   = (const float*)d_in[34];
  const float* emS_in   = (const float*)d_in[35];
  const int*   ids      = (const int*)d_in[36];
  float* logits = (float*)d_out;

  // ---- workspace carve-out ----
  float* ws = (float*)d_ws;
  size_t off = 0;
  auto alloc = [&](size_t n){ float* p = ws + off; off += n; return p; };
  float* x     = alloc((size_t)TT*DMODEL);   // reused as xf later
  float* xcols = alloc((size_t)TT*GD);
  float* xout  = alloc((size_t)TT*GD);
  float* kc    = alloc((size_t)TT*GD);
  float* vc    = alloc((size_t)TT*GD);
  float* qn    = alloc((size_t)TT*GD);
  float* vn    = alloc((size_t)TT*GD);
  float* zh0   = alloc((size_t)TT*GD);
  float* zh1   = alloc((size_t)TT*GD);
  float* novb  = alloc((size_t)BSZ*NB*SEQ*NC);
  float* gate  = alloc((size_t)TT*GG);
  float* gww   = alloc((size_t)BSZ*NB*SEQ*NC*RS);
  float* eligK = alloc((size_t)BSZ*NB*RS*DMEM);
  float* eligV = alloc((size_t)BSZ*NB*RS*DMEM);
  float* pmK   = alloc((size_t)BSZ*NB*RS*DMEM);
  float* pmV   = alloc((size_t)BSZ*NB*RS*DMEM);
  float* pma   = alloc((size_t)BSZ*NB*RS);
  float* emK   = alloc((size_t)BSZ*NB*MEM*DMEM);
  float* emV   = alloc((size_t)BSZ*NB*MEM*DMEM);
  float* emS   = alloc((size_t)BSZ*NB*MEM);
  float* tops  = alloc((size_t)BSZ*NB*CEM);
  int*   topi  = (int*)alloc((size_t)BSZ*NB*CEM);

  // ---- copy mutable state (never mutate inputs) ----
  hipMemcpyAsync(pmK, pmK_in, sizeof(float)*BSZ*NB*RS*DMEM, hipMemcpyDeviceToDevice, stream);
  hipMemcpyAsync(pmV, pmV_in, sizeof(float)*BSZ*NB*RS*DMEM, hipMemcpyDeviceToDevice, stream);
  hipMemcpyAsync(pma, pma_in, sizeof(float)*BSZ*NB*RS,      hipMemcpyDeviceToDevice, stream);
  hipMemcpyAsync(emK, emK_in, sizeof(float)*BSZ*NB*MEM*DMEM,hipMemcpyDeviceToDevice, stream);
  hipMemcpyAsync(emV, emV_in, sizeof(float)*BSZ*NB*MEM*DMEM,hipMemcpyDeviceToDevice, stream);
  hipMemcpyAsync(emS, emS_in, sizeof(float)*BSZ*NB*MEM,     hipMemcpyDeviceToDevice, stream);

  // ---- forward ----
  embed_kernel<<<TT, 256, 0, stream>>>(emb, pos_emb, ids, x);

  // x_cols = x @ W_fo + b_fo   [2048 x 768] @ [768 x 2048]
  gemm_wmma_kernel<DMODEL, GD, false, true><<<dim3(GD/64, TT/64), 128, 0, stream>>>(
      x, W_fo, b_fo, xcols);

  for (int r = 0; r < 3; ++r) {
    columns_kernel<<<dim3(GG, TT), 64, 0, stream>>>(
        xcols, W_col, b_col, W_q, pmK, pmV, emK, emV, Wback_pm, Wback_em, xout);
    float* zh_out  = (r & 1) ? zh1 : zh0;
    float* zh_prev = (r & 1) ? zh0 : zh1;
    heads_kernel<<<dim3(GG, TT), 64, 0, stream>>>(
        xout, W_k, W_v, W_q, W_vn, w_gate, w_wnov, W_pred, emK, zh_prev,
        (r == 0) ? 1 : 0, kc, vc, qn, vn, zh_out, gate, novb);
    route_kernel<<<dim3(GG, TT), 32, 0, stream>>>(kc, pmK, gate, gww);
    elig_kernel<<<dim3(RS, BSZ*NB), 64, 0, stream>>>(gww, kc, vc, eligK, eligV);
    topk_kernel<<<BSZ*NB, 64, 0, stream>>>(novb, tops, topi);
    pm_commit_kernel<<<BSZ*NB, 64, 0, stream>>>(
        eligK, eligV, W1_pm, b1_pm, Wg_pm, Ws_pm, Wt_pm, pmK, pmV, pma);
    em_write_kernel<<<BSZ*NB, 64, 0, stream>>>(
        qn, vn, tops, topi, W1_em, b1_em, Wg_em, Wt_em, Wd_em, emK, emV, emS);
    mix_kernel<<<((size_t)TT*GD + 255)/256, 256, 0, stream>>>(
        xcols, xout, lam_l, (r == 0) ? 1 : 0);
  }

  // xf = x_cols @ W_fi + b_fi   (reuse x buffer)
  float* xf = x;
  gemm_wmma_kernel<GD, DMODEL, false, true><<<dim3(DMODEL/64, TT/64), 128, 0, stream>>>(
      xcols, W_fi, b_fi, xf);
  layernorm_kernel<<<TT, 256, 0, stream>>>(xf, ln_g, ln_b);

  // logits = xf @ emb^T   [2048 x 768] @ [768 x 32000]
  gemm_wmma_kernel<DMODEL, VOCAB, true, false><<<dim3(VOCAB/64, TT/64), 128, 0, stream>>>(
      xf, emb, nullptr, logits);
}
